// SpikingNet_37297495998527
// MI455X (gfx1250) — compile-verified, matrix operand order checked by reading, not run on hardware
//
#include <hip/hip_runtime.h>
#include <stdint.h>

// ---------------------------------------------------------------------------
// Spiking net forward on gfx1250.
//  - cur1 = x@W1^T hoisted out of the scan into one full-chip f16 WMMA GEMM
//    (8000 blocks) when workspace allows; fused per-step fallback otherwise.
//  - per-step kernel: recurrent GEMM (spk@Wrec^T) on v_wmma_f32_16x16x32_f16,
//    LIF elementwise in WMMA C-register layout, layer-2 GEMM fused.
//  - previous-spike tile restage uses global_load_async_to_lds_b128 (ASYNCcnt).
// ---------------------------------------------------------------------------

typedef __attribute__((ext_vector_type(16))) _Float16 v16h;
typedef __attribute__((ext_vector_type(8)))  _Float16 v8h;
typedef __attribute__((ext_vector_type(8)))  float    v8f;

static constexpr int kT    = 250;
static constexpr int kB    = 512;
static constexpr int kI    = 700;
static constexpr int kIP   = 704;   // I padded to multiple of 32
static constexpr int kH    = 1024;
static constexpr int kO    = 20;
static constexpr int kOP   = 32;    // O padded to 32 (2 WMMA N-tiles)
static constexpr int kBM   = 16;    // batch rows per block
static constexpr int kLDA  = 1040;  // LDS row stride (halves), 16B-aligned rows

union Frag {
    v16h v;
    v8h  h[2];
};

__device__ __forceinline__ v8f wmma_step(const Frag& a, const Frag& b, v8f c) {
    // v_wmma_f32_16x16x32_f16  D = A(16x32 f16) x B(32x16 f16) + C(16x16 f32)
    return __builtin_amdgcn_wmma_f32_16x16x32_f16(
        false, a.v, false, b.v, (short)0, c, false, false);
}

__device__ __forceinline__ void load_frag(Frag& f, const _Float16* p) {
    f.h[0] = *(const v8h*)(p);
    f.h[1] = *(const v8h*)(p + 16);
}

// async global->LDS 16B copy, tracked by ASYNCcnt (ISA 10.x / 15.18.3)
__device__ __forceinline__ void async_copy_b128(const _Float16* lds_dst,
                                                const _Float16* gsrc) {
    const uint32_t lds_off = (uint32_t)(uintptr_t)lds_dst;  // addr[31:0] = LDS offset
    asm volatile("global_load_async_to_lds_b128 %0, %1, off"
                 :
                 : "v"(lds_off), "v"((uint64_t)(uintptr_t)gsrc)
                 : "memory");
}
__device__ __forceinline__ void wait_async0() {
    asm volatile("s_wait_asynccnt 0x0" ::: "memory");
}

// --- one-time weight convert/pad to f16 -------------------------------------
__global__ __launch_bounds__(256) void convert_weights_kernel(
    const float* __restrict__ W1,    // [H][I]   fp32
    const float* __restrict__ Wrec,  // [H][H]   fp32
    const float* __restrict__ W2,    // [O][H]   fp32
    _Float16* __restrict__ W1h,      // [H][IP]  f16, zero-padded cols
    _Float16* __restrict__ Wrech,    // [H][H]   f16
    _Float16* __restrict__ W2h)      // [OP][H]  f16, zero-padded rows
{
    const int idx = blockIdx.x * 256 + threadIdx.x;
    if (idx < kH * kIP) {
        const int h = idx / kIP;
        const int k = idx - h * kIP;
        W1h[idx] = (_Float16)((k < kI) ? W1[(size_t)h * kI + k] : 0.0f);
    }
    if (idx < kH * kH) {
        Wrech[idx] = (_Float16)Wrec[idx];
    }
    if (idx < kOP * kH) {
        W2h[idx] = (_Float16)((idx < kO * kH) ? W2[idx] : 0.0f);
    }
}

// --- full-chip input GEMM: cur1[T*B][H] = x[T*B][I] @ W1^T -------------------
__global__ __launch_bounds__(256) void cur1_gemm_kernel(
    const float*    __restrict__ x,     // [T*B][I] fp32
    const _Float16* __restrict__ W1h,   // [H][IP]
    _Float16*       __restrict__ cur1h) // [T*B][H] f16
{
    __shared__ __align__(16) _Float16 As[kBM * kLDA];

    const int tid   = threadIdx.x;
    const int lane  = tid & 31;
    const int wave  = tid >> 5;
    const int row0  = blockIdx.x * kBM;

    const int mrow  = lane & 15;
    const int khalf = (lane < 16) ? 0 : 8;
    const int mbase = (lane < 16) ? 0 : 8;
    const int n0    = wave * 128;

    for (int idx = tid; idx < kBM * kIP; idx += 256) {
        const int r = idx / kIP;
        const int c = idx - r * kIP;
        const float v = (c < kI) ? x[(size_t)(row0 + r) * kI + c] : 0.0f;
        As[r * kLDA + c] = (_Float16)v;
    }
    __syncthreads();

    v8f acc[8];
#pragma unroll
    for (int j = 0; j < 8; ++j) acc[j] = (v8f){0, 0, 0, 0, 0, 0, 0, 0};

    for (int kc = 0; kc < kIP / 32; ++kc) {
        Frag af;
        load_frag(af, &As[mrow * kLDA + kc * 32 + khalf]);
#pragma unroll
        for (int nt = 0; nt < 8; ++nt) {
            const int nrow = n0 + nt * 16 + (lane & 15);
            Frag bf;
            load_frag(bf, &W1h[(size_t)nrow * kIP + kc * 32 + khalf]);
            acc[nt] = wmma_step(af, bf, acc[nt]);
        }
    }

#pragma unroll
    for (int nt = 0; nt < 8; ++nt) {
        const int n = n0 + nt * 16 + (lane & 15);
#pragma unroll
        for (int v = 0; v < 8; ++v) {
            const int m = v + mbase;
            cur1h[(size_t)(row0 + m) * kH + n] = (_Float16)acc[nt][v];
        }
    }
}

// --- fused one-timestep kernel ----------------------------------------------
// PRECUR=true : cur1 read from precomputed f16 tensor (input GEMM hoisted)
// PRECUR=false: cur1 GEMM fused here (workspace-constrained fallback)
template <bool PRECUR>
__global__ __launch_bounds__(256) void snn_step_kernel(
    const float*    __restrict__ xt,      // [B][I] fp32 (unused if PRECUR)
    const _Float16* __restrict__ cur1t,   // [B][H] f16  (unused if !PRECUR)
    const _Float16* __restrict__ W1h,     // [H][IP]
    const _Float16* __restrict__ Wrech,   // [H][H]
    const _Float16* __restrict__ W2h,     // [OP][H]
    _Float16*       __restrict__ spk1,    // [B][H] state (exact 0/1 in f16)
    float*          __restrict__ syn1,    // [B][H] state
    float*          __restrict__ mem1,    // [B][H] state
    float*          __restrict__ mem2,    // [B][O] state
    float*          __restrict__ spk2_out,   // [B][O] slice of d_out
    float*          __restrict__ mem2_out,   // [B][O] slice of d_out
    const float*    __restrict__ alpha_p,
    const float*    __restrict__ beta1_p,
    const float*    __restrict__ th1_p)
{
    __shared__ __align__(16) _Float16 As[kBM * kLDA];

    const int tid  = threadIdx.x;
    const int lane = tid & 31;
    const int wave = tid >> 5;
    const int row0 = blockIdx.x * kBM;

    const float a_c = fminf(fmaxf(alpha_p[0], 0.0f), 1.0f);
    const float b_c = fminf(fmaxf(beta1_p[0], 0.0f), 1.0f);
    const float th1 = th1_p[0];

    // WMMA fragment lane mapping (ISA 16-bit 16x32 layout):
    //   lane<16 holds K in {0..7, 16..23}, lane>=16 holds K in {8..15, 24..31}
    const int mrow  = lane & 15;
    const int khalf = (lane < 16) ? 0 : 8;
    const int mbase = (lane < 16) ? 0 : 8;   // C-layout row base per half-wave
    const int n0    = wave * 128;            // each wave owns 128 N-columns

    v8f acc[8];
#pragma unroll
    for (int j = 0; j < 8; ++j) acc[j] = (v8f){0, 0, 0, 0, 0, 0, 0, 0};

    if constexpr (!PRECUR) {
        // ---- stage xt tile (fp32 -> f16, pad to kIP) ------------------------
        for (int idx = tid; idx < kBM * kIP; idx += 256) {
            const int r = idx / kIP;
            const int c = idx - r * kIP;
            const float v = (c < kI) ? xt[(size_t)(row0 + r) * kI + c] : 0.0f;
            As[r * kLDA + c] = (_Float16)v;
        }
        __syncthreads();

        // ---- GEMM 1: cur1 = xt @ W1^T  (K = 704) ----------------------------
        for (int kc = 0; kc < kIP / 32; ++kc) {
            Frag af;
            load_frag(af, &As[mrow * kLDA + kc * 32 + khalf]);
#pragma unroll
            for (int nt = 0; nt < 8; ++nt) {
                const int nrow = n0 + nt * 16 + (lane & 15);
                Frag bf;
                load_frag(bf, &W1h[(size_t)nrow * kIP + kc * 32 + khalf]);
                acc[nt] = wmma_step(af, bf, acc[nt]);
            }
        }
        __syncthreads();
    }

    // ---- stage previous spikes via async global->LDS b128 copies ------------
    // 16 rows x 1024 f16 = 2048 B/row; 128 lanes x 16 B cover one row.
    {
        const int lane128 = tid & 127;
        const int rbase   = tid >> 7;     // 0..1
#pragma unroll
        for (int it = 0; it < 8; ++it) {
            const int r = rbase + it * 2;
            async_copy_b128(&As[r * kLDA + lane128 * 8],
                            spk1 + (size_t)(row0 + r) * kH + lane128 * 8);
        }
        wait_async0();
    }
    __syncthreads();

    // ---- GEMM 2: += spk_prev @ Wrec^T  (K = 1024) ---------------------------
    for (int kc = 0; kc < kH / 32; ++kc) {
        Frag af;
        load_frag(af, &As[mrow * kLDA + kc * 32 + khalf]);
#pragma unroll
        for (int nt = 0; nt < 8; ++nt) {
            const int nrow = n0 + nt * 16 + (lane & 15);
            Frag bf;
            load_frag(bf, &Wrech[(size_t)nrow * kH + kc * 32 + khalf]);
            acc[nt] = wmma_step(af, bf, acc[nt]);
        }
    }
    __syncthreads();   // all As reads done; reuse As for new spikes

    // ---- layer-1 LIF elementwise, directly in WMMA C layout -----------------
    // C layout: vgpr v, lane l -> m = v + (l<16?0:8), n = tile_n0 + (l&15)
#pragma unroll
    for (int nt = 0; nt < 8; ++nt) {
        const int n = n0 + nt * 16 + (lane & 15);
#pragma unroll
        for (int v = 0; v < 8; ++v) {
            const int m = v + mbase;
            const size_t g = (size_t)(row0 + m) * kH + n;
            const float cin   = PRECUR ? (float)cur1t[g] : 0.0f;
            const float syn_o = syn1[g];
            const float mem_o = mem1[g];
            const float dr    = (mem_o - th1 > 0.0f) ? 1.0f : 0.0f;
            const float syn_n = a_c * syn_o + acc[nt][v] + cin;
            float       mem_n = b_c * mem_o + syn_n - dr * th1;
            const float spk   = (mem_n - th1 > 0.0f) ? 1.0f : 0.0f;
            mem_n             = mem_n - (spk - dr) * th1;
            syn1[g] = syn_n;
            mem1[g] = mem_n;
            const _Float16 sh = (_Float16)spk;
            spk1[g] = sh;            // state for next timestep
            As[m * kLDA + n] = sh;   // feed layer-2 GEMM
        }
    }
    __syncthreads();

    // ---- layer 2: cur2 = spk_new @ W2^T (N = 32 padded, waves 0..1) ---------
    if (wave < 2) {
        v8f acc2 = (v8f){0, 0, 0, 0, 0, 0, 0, 0};
        for (int kc = 0; kc < kH / 32; ++kc) {
            Frag af;
            load_frag(af, &As[mrow * kLDA + kc * 32 + khalf]);
            const int nrow = wave * 16 + (lane & 15);
            Frag bf;
            load_frag(bf, &W2h[(size_t)nrow * kH + kc * 32 + khalf]);
            acc2 = wmma_step(af, bf, acc2);
        }
        // Leaky layer-2 epilogue (beta2 = 0.9, th2 = 1.0, subtract reset)
#pragma unroll
        for (int v = 0; v < 8; ++v) {
            const int m = v + mbase;
            const int n = wave * 16 + (lane & 15);
            if (n < kO) {
                const size_t g = (size_t)(row0 + m) * kO + n;
                const float mo = mem2[g];
                const float r2 = (mo - 1.0f > 0.0f) ? 1.0f : 0.0f;
                const float mn = 0.9f * mo + acc2[v] - r2;
                const float s2 = (mn - 1.0f > 0.0f) ? 1.0f : 0.0f;
                mem2[g]     = mn;
                spk2_out[g] = s2;
                mem2_out[g] = mn;
            }
        }
    }
}

// ---------------------------------------------------------------------------
static inline size_t align256(size_t x) { return (x + 255) & ~(size_t)255; }

extern "C" void kernel_launch(void* const* d_in, const int* in_sizes, int n_in,
                              void* d_out, int out_size, void* d_ws, size_t ws_size,
                              hipStream_t stream) {
    (void)in_sizes; (void)n_in; (void)out_size;

    const float* x     = (const float*)d_in[0];   // [T,B,I]
    const float* W1    = (const float*)d_in[1];   // [H,I]
    const float* Wrec  = (const float*)d_in[2];   // [H,H]
    const float* W2    = (const float*)d_in[3];   // [O,H]
    const float* alpha = (const float*)d_in[4];
    const float* beta1 = (const float*)d_in[5];
    const float* th1   = (const float*)d_in[6];
    float* out = (float*)d_out;                   // spk2[T,B,O] ++ mem2[T,B,O]

    // workspace carve-up
    char* base = (char*)d_ws;
    char* ws   = base;
    _Float16* W1h   = (_Float16*)ws; ws += align256((size_t)kH * kIP * sizeof(_Float16));
    _Float16* Wrech = (_Float16*)ws; ws += align256((size_t)kH * kH  * sizeof(_Float16));
    _Float16* W2h   = (_Float16*)ws; ws += align256((size_t)kOP * kH * sizeof(_Float16));
    char* state = ws;
    _Float16* spk1 = (_Float16*)ws;  ws += align256((size_t)kB * kH * sizeof(_Float16));
    float* syn1    = (float*)ws;     ws += align256((size_t)kB * kH * sizeof(float));
    float* mem1    = (float*)ws;     ws += align256((size_t)kB * kH * sizeof(float));
    float* mem2    = (float*)ws;     ws += align256((size_t)kB * kO * sizeof(float));
    const size_t state_bytes = (size_t)(ws - state);

    _Float16* cur1h = (_Float16*)ws;
    const size_t cur1_bytes = (size_t)kT * kB * kH * sizeof(_Float16);  // 256 MB
    const bool precur = ((size_t)(ws - base) + cur1_bytes) <= ws_size;

    // zero recurrent state every call (harness poisons ws once, never restores)
    hipMemsetAsync(state, 0, state_bytes, stream);

    // weight conversion (deterministic, redone each call)
    convert_weights_kernel<<<(kH * kH + 255) / 256, 256, 0, stream>>>(
        W1, Wrec, W2, W1h, Wrech, W2h);

    if (precur) {
        // hoisted full-chip input GEMM over all T*B rows
        cur1_gemm_kernel<<<(kT * kB) / kBM, 256, 0, stream>>>(x, W1h, cur1h);
    }

    // sequential scan over timesteps
    const size_t outHalf = (size_t)kT * kB * kO;
    for (int t = 0; t < kT; ++t) {
        if (precur) {
            snn_step_kernel<true><<<kB / kBM, 256, 0, stream>>>(
                nullptr, cur1h + (size_t)t * kB * kH,
                W1h, Wrech, W2h, spk1, syn1, mem1, mem2,
                out + (size_t)t * kB * kO,
                out + outHalf + (size_t)t * kB * kO,
                alpha, beta1, th1);
        } else {
            snn_step_kernel<false><<<kB / kBM, 256, 0, stream>>>(
                x + (size_t)t * kB * kI, nullptr,
                W1h, Wrech, W2h, spk1, syn1, mem1, mem2,
                out + (size_t)t * kB * kO,
                out + outHalf + (size_t)t * kB * kO,
                alpha, beta1, th1);
        }
    }
}
